// MultiHeadAttention_90744069030376
// MI455X (gfx1250) — matrix operand order repaired and checked by reading, not compiled, not run on hardware
//
#include <hip/hip_runtime.h>
#include <hip/hip_bf16.h>
#include <cstdint>

// MHA forward for MI455X (gfx1250, wave32).
// - All matmuls: v_wmma_f32_16x16x32_bf16, fp32 accumulation.
// - Intermediates (qp/kp/vp, feat_ctx) stored bf16 in global: halves the
//   intermediate HBM traffic and lets tiles stage into LDS with the CDNA5
//   async path (global_load_async_to_lds_b128 + s_wait_asynccnt).
// - The 1GB fp32 attn materialization dominates bytes; everything else is
//   arranged so compute and staging hide under it.

#define Bc 4
#define Sc 2048
#define Dc 1024
#define Ec 1024
#define Hc 16
#define DHc 64

typedef __attribute__((ext_vector_type(16))) __bf16 v16bf;
typedef __attribute__((ext_vector_type(8)))  __bf16 v8bf;
typedef __attribute__((ext_vector_type(4)))  __bf16 v4bf;
typedef __attribute__((ext_vector_type(8)))  float  v8f;

static __device__ inline __bf16 f2bf(float f) {
  union { float f; unsigned int u; } v; v.f = f;
  unsigned int r = v.u + 0x7FFFu + ((v.u >> 16) & 1u);   // round-to-nearest-even
  return __builtin_bit_cast(__bf16, (unsigned short)(r >> 16));
}

static __device__ inline v8f vzero() {
  v8f z;
#pragma unroll
  for (int i = 0; i < 8; ++i) z[i] = 0.0f;
  return z;
}

// Generic pointer to LDS byte offset (LDS generic ptrs carry the offset in
// the low 32 bits on amdgcn).
static __device__ inline unsigned lds_off(const void* p) {
  return (unsigned)(uintptr_t)p;
}

// CDNA5 async global->LDS copy, 16B per lane. Tracked by ASYNCcnt.
static __device__ inline void async_ld_b128(unsigned lds_byte_off,
                                            const void* gaddr) {
  asm volatile("global_load_async_to_lds_b128 %0, %1, off"
               :
               : "v"(lds_byte_off), "v"((unsigned long long)gaddr)
               : "memory");
}
static __device__ inline void wait_async() {
  asm volatile("s_wait_asynccnt 0" ::: "memory");
}

// A-fragment (16x32 bf16, MxK) per CDNA5 ISA layout:
// lane<16: M=lane, K = kbase+{0..7} and kbase+{16..23}
// lane>=16: M=lane-16, K = kbase+{8..15} and kbase+{24..31}
static __device__ inline v16bf load_a(const __bf16* base, int stride, int kbase, int lane) {
  const int m    = lane & 15;
  const int koff = kbase + ((lane >> 4) << 3);
  const __bf16* p = base + m * stride + koff;
  v8bf lo = *(const v8bf*)p;
  v8bf hi = *(const v8bf*)(p + 16);
  v16bf r;
#pragma unroll
  for (int i = 0; i < 8; ++i) { r[i] = lo[i]; r[i + 8] = hi[i]; }
  return r;
}

// B-fragment (32x16 bf16, KxN), LDS staged N-major (Bt[n][k]):
// lane<16: N=lane, K = kbase+{0..7},{16..23} ; lane>=16: N=lane-16, K = kbase+{8..15},{24..31}
static __device__ inline v16bf load_b(const __bf16* base, int stride, int kbase, int lane) {
  const int n    = lane & 15;
  const int koff = kbase + ((lane >> 4) << 3);
  const __bf16* p = base + n * stride + koff;
  v8bf lo = *(const v8bf*)p;
  v8bf hi = *(const v8bf*)(p + 16);
  v16bf r;
#pragma unroll
  for (int i = 0; i < 8; ++i) { r[i] = lo[i]; r[i + 8] = hi[i]; }
  return r;
}

static __device__ inline v8f wmma_bf16(v16bf a, v16bf b, v8f c) {
  return __builtin_amdgcn_wmma_f32_16x16x32_bf16(false, a, false, b, (short)0, c,
                                                 false, false);
}

// ---------------------------------------------------------------------------
// C[M,N] = A[M,K] @ W[K,N] + bias  (W fp32; A fp32 or bf16; C fp32 or bf16)
// 128x128 tile, BK=32, 256 threads = 8 waves in a 4x2 wave grid.
// ---------------------------------------------------------------------------
template <typename TA, bool BF16OUT>
__global__ __launch_bounds__(256) void gemm_bias_kernel(
    const TA* __restrict__ A, const float* __restrict__ W,
    const float* __restrict__ bias, void* __restrict__ Cout,
    int M, int N, int K) {
  __shared__ __align__(16) __bf16 As[128 * 32];
  __shared__ __align__(16) __bf16 Bs[128 * 32];  // Bt[n][k]
  const int tid = threadIdx.x, lane = tid & 31, wave = tid >> 5;
  const int bm = blockIdx.x * 128, bn = blockIdx.y * 128;
  const int wm = (wave >> 1) * 32, wn = (wave & 1) * 64;

  v8f acc[2][4];
#pragma unroll
  for (int mi = 0; mi < 2; ++mi)
#pragma unroll
    for (int ni = 0; ni < 4; ++ni) acc[mi][ni] = vzero();

  for (int k0 = 0; k0 < K; k0 += 32) {
    // ---- stage A (128 x 32) ----
    if constexpr (sizeof(TA) == 2) {
      // bf16 A: pure async DMA, no VALU work. 512 x 16B chunks.
      for (int c = tid; c < 512; c += 256) {
        const int i = c * 8, r = i >> 5, k = i & 31;
        async_ld_b128(lds_off(&As[i]), &A[(size_t)(bm + r) * K + k0 + k]);
      }
    } else {
      // fp32 A: float4 load, convert, 8B packed LDS store.
      for (int c = tid; c < 1024; c += 256) {
        const int i = c * 4, r = i >> 5, k = i & 31;
        const float4 f = *(const float4*)&A[(size_t)(bm + r) * K + k0 + k];
        v4bf o; o[0] = f2bf(f.x); o[1] = f2bf(f.y); o[2] = f2bf(f.z); o[3] = f2bf(f.w);
        *(v4bf*)&As[i] = o;
      }
    }
    // ---- stage W transposed (Bt[n][k]), float4 along n ----
    for (int c = tid; c < 1024; c += 256) {
      const int i = c * 4, n = i & 127, k = i >> 7;
      const float4 f = *(const float4*)&W[(size_t)(k0 + k) * N + bn + n];
      Bs[(n + 0) * 32 + k] = f2bf(f.x);
      Bs[(n + 1) * 32 + k] = f2bf(f.y);
      Bs[(n + 2) * 32 + k] = f2bf(f.z);
      Bs[(n + 3) * 32 + k] = f2bf(f.w);
    }
    if constexpr (sizeof(TA) == 2) wait_async();
    __syncthreads();

    v16bf af[2], bf_[4];
#pragma unroll
    for (int mi = 0; mi < 2; ++mi) af[mi] = load_a(As + (wm + mi * 16) * 32, 32, 0, lane);
#pragma unroll
    for (int ni = 0; ni < 4; ++ni) bf_[ni] = load_b(Bs + (wn + ni * 16) * 32, 32, 0, lane);
#pragma unroll
    for (int mi = 0; mi < 2; ++mi)
#pragma unroll
      for (int ni = 0; ni < 4; ++ni) acc[mi][ni] = wmma_bf16(af[mi], bf_[ni], acc[mi][ni]);
    __syncthreads();
  }

  const int rofs = (lane >> 4) << 3, col0 = lane & 15;
#pragma unroll
  for (int mi = 0; mi < 2; ++mi)
#pragma unroll
    for (int ni = 0; ni < 4; ++ni) {
      const int col = bn + wn + ni * 16 + col0;
      const float bv = bias[col];
#pragma unroll
      for (int r = 0; r < 8; ++r) {
        const int row = bm + wm + mi * 16 + rofs + r;
        const float val = acc[mi][ni][r] + bv;
        if constexpr (BF16OUT)
          ((__bf16*)Cout)[(size_t)row * N + col] = f2bf(val);
        else
          ((float*)Cout)[(size_t)row * N + col] = val;
      }
    }
}

// ---------------------------------------------------------------------------
// scores[h*B+b, q, k] = scale * dot(qp[b,q,h,:], kp[b,k,h,:])   (DH = 64)
// qp/kp are bf16: Q/K tiles staged purely via async global->LDS.
// grid: x = k-tile (16), y = q-tile (16), z = b*H + h (64). 128x128 tile.
// ---------------------------------------------------------------------------
__global__ __launch_bounds__(256) void attn_scores_kernel(
    const __bf16* __restrict__ qp, const __bf16* __restrict__ kp,
    float* __restrict__ attn) {
  __shared__ __align__(16) __bf16 Qs[128 * 64];
  __shared__ __align__(16) __bf16 Ks[128 * 64];  // Kt[n][d] (= kp rows directly)
  const int tid = threadIdx.x, lane = tid & 31, wave = tid >> 5;
  const int bk = blockIdx.x * 128, bq = blockIdx.y * 128;
  const int bh = blockIdx.z, b = bh >> 4, h = bh & 15;
  const int wm = (wave >> 1) * 32, wn = (wave & 1) * 64;

  const __bf16* qbase = qp + (size_t)(b * Sc + bq) * Ec + h * DHc;
  const __bf16* kbase = kp + (size_t)(b * Sc + bk) * Ec + h * DHc;
  for (int c = tid; c < 1024; c += 256) {      // 256 rows x 4 chunks of 8 bf16
    const int r = c >> 2, d8 = (c & 3) * 8;
    async_ld_b128(lds_off(&Qs[r * 64 + d8]), qbase + (size_t)r * Ec + d8);
    async_ld_b128(lds_off(&Ks[r * 64 + d8]), kbase + (size_t)r * Ec + d8);
  }
  wait_async();
  __syncthreads();

  v8f acc[2][4];
#pragma unroll
  for (int mi = 0; mi < 2; ++mi)
#pragma unroll
    for (int ni = 0; ni < 4; ++ni) acc[mi][ni] = vzero();

#pragma unroll
  for (int ks = 0; ks < 64; ks += 32) {
    v16bf a0 = load_a(Qs + (wm + 0) * 64, 64, ks, lane);
    v16bf a1 = load_a(Qs + (wm + 16) * 64, 64, ks, lane);
#pragma unroll
    for (int ni = 0; ni < 4; ++ni) {
      v16bf bf_ = load_b(Ks + (wn + ni * 16) * 64, 64, ks, lane);
      acc[0][ni] = wmma_bf16(a0, bf_, acc[0][ni]);
      acc[1][ni] = wmma_bf16(a1, bf_, acc[1][ni]);
    }
  }

  const float scale = 0.125f;  // 1/sqrt(64)
  const int rofs = (lane >> 4) << 3, col0 = lane & 15;
  const size_t rowbase = (size_t)(h * Bc + b) * Sc;
#pragma unroll
  for (int mi = 0; mi < 2; ++mi)
#pragma unroll
    for (int ni = 0; ni < 4; ++ni) {
      const int col = bk + wn + ni * 16 + col0;
#pragma unroll
      for (int r = 0; r < 8; ++r) {
        const size_t row = rowbase + (bq + wm + mi * 16 + rofs + r);
        attn[row * Sc + col] = acc[mi][ni][r] * scale;
      }
    }
}

// ---------------------------------------------------------------------------
// In-place softmax over rows of 2048. One block per row (H*B*S rows).
// ---------------------------------------------------------------------------
__global__ __launch_bounds__(256) void softmax_kernel(float* __restrict__ attn) {
  __shared__ float red[256];
  const int tid = threadIdx.x;
  float* p = attn + (size_t)blockIdx.x * Sc;
  float4 v[2];
  float m = -1e30f;
#pragma unroll
  for (int j = 0; j < 2; ++j) {
    v[j] = *(const float4*)&p[(tid + j * 256) * 4];
    m = fmaxf(m, fmaxf(fmaxf(v[j].x, v[j].y), fmaxf(v[j].z, v[j].w)));
  }
  red[tid] = m; __syncthreads();
  for (int st = 128; st > 0; st >>= 1) {
    if (tid < st) red[tid] = fmaxf(red[tid], red[tid + st]);
    __syncthreads();
  }
  m = red[0]; __syncthreads();
  float s = 0.0f;
#pragma unroll
  for (int j = 0; j < 2; ++j) {
    v[j].x = __expf(v[j].x - m); v[j].y = __expf(v[j].y - m);
    v[j].z = __expf(v[j].z - m); v[j].w = __expf(v[j].w - m);
    s += v[j].x + v[j].y + v[j].z + v[j].w;
  }
  red[tid] = s; __syncthreads();
  for (int st = 128; st > 0; st >>= 1) {
    if (tid < st) red[tid] += red[tid + st];
    __syncthreads();
  }
  const float inv = 1.0f / red[0];
#pragma unroll
  for (int j = 0; j < 2; ++j) {
    float4 o; o.x = v[j].x * inv; o.y = v[j].y * inv;
    o.z = v[j].z * inv; o.w = v[j].w * inv;
    *(float4*)&p[(tid + j * 256) * 4] = o;
  }
}

// ---------------------------------------------------------------------------
// feat_ctx[b, q, h*64+d] = sum_k attn[h*B+b, q, k] * vp[b, k, h*64+d]
// attn fp32 (must convert), vp bf16. Output feat_ctx bf16.
// grid: y = q-tile (16), z = b*H + h (64). 128(q) x 64(d) tile, K over S.
// ---------------------------------------------------------------------------
__global__ __launch_bounds__(256) void attn_v_kernel(
    const float* __restrict__ attn, const __bf16* __restrict__ vp,
    __bf16* __restrict__ fctx) {
  __shared__ __align__(16) __bf16 As[128 * 32];
  __shared__ __align__(16) __bf16 Vs[64 * 32];   // Vt[n][k]
  const int tid = threadIdx.x, lane = tid & 31, wave = tid >> 5;
  const int bq = blockIdx.y * 128;
  const int bh = blockIdx.z, b = bh >> 4, h = bh & 15;
  const size_t arow = (size_t)(h * Bc + b) * Sc;

  v8f acc[4];
#pragma unroll
  for (int ni = 0; ni < 4; ++ni) acc[ni] = vzero();

  for (int k0 = 0; k0 < Sc; k0 += 32) {
    for (int c = tid; c < 1024; c += 256) {      // attn: float4 -> 4x bf16
      const int i = c * 4, r = i >> 5, k = i & 31;
      const float4 f = *(const float4*)&attn[(arow + bq + r) * Sc + k0 + k];
      v4bf o; o[0] = f2bf(f.x); o[1] = f2bf(f.y); o[2] = f2bf(f.z); o[3] = f2bf(f.w);
      *(v4bf*)&As[i] = o;
    }
    for (int c = tid; c < 256; c += 256) {       // V: 8-wide bf16, transpose in LDS
      const int i = c * 8, k = i >> 6, n = i & 63;
      v8bf v = *(const v8bf*)&vp[(size_t)(b * Sc + k0 + k) * Ec + h * DHc + n];
#pragma unroll
      for (int j = 0; j < 8; ++j) Vs[(n + j) * 32 + k] = v[j];
    }
    __syncthreads();
    v16bf a = load_a(As + (wave * 16) * 32, 32, 0, lane);
#pragma unroll
    for (int ni = 0; ni < 4; ++ni) {
      v16bf bf_ = load_b(Vs + (ni * 16) * 32, 32, 0, lane);
      acc[ni] = wmma_bf16(a, bf_, acc[ni]);
    }
    __syncthreads();
  }

  const int rofs = (lane >> 4) << 3, col0 = lane & 15;
#pragma unroll
  for (int ni = 0; ni < 4; ++ni)
#pragma unroll
    for (int r = 0; r < 8; ++r) {
      const int row = bq + wave * 16 + rofs + r;
      fctx[(size_t)(b * Sc + row) * Ec + h * DHc + ni * 16 + col0] = f2bf(acc[ni][r]);
    }
}

// ---------------------------------------------------------------------------
// out = LayerNorm(fo + q) * gamma + beta. One block per row of 1024.
// ---------------------------------------------------------------------------
__global__ __launch_bounds__(256) void resid_ln_kernel(
    const float* __restrict__ fo, const float* __restrict__ qres,
    const float* __restrict__ gamma, const float* __restrict__ beta,
    float* __restrict__ out) {
  __shared__ float r1[256];
  __shared__ float r2[256];
  const int tid = threadIdx.x;
  const size_t row = blockIdx.x;
  const float4 a = *(const float4*)&fo[row * Dc + tid * 4];
  const float4 q = *(const float4*)&qres[row * Dc + tid * 4];
  float4 x; x.x = a.x + q.x; x.y = a.y + q.y; x.z = a.z + q.z; x.w = a.w + q.w;
  const float s  = x.x + x.y + x.z + x.w;
  const float s2 = x.x * x.x + x.y * x.y + x.z * x.z + x.w * x.w;
  r1[tid] = s; r2[tid] = s2; __syncthreads();
  for (int st = 128; st > 0; st >>= 1) {
    if (tid < st) { r1[tid] += r1[tid + st]; r2[tid] += r2[tid + st]; }
    __syncthreads();
  }
  const float mean = r1[0] * (1.0f / Dc);
  const float var  = r2[0] * (1.0f / Dc) - mean * mean;
  const float inv  = rsqrtf(var + 1e-5f);
  const float4 g = *(const float4*)&gamma[tid * 4];
  const float4 bt = *(const float4*)&beta[tid * 4];
  float4 o;
  o.x = (x.x - mean) * inv * g.x + bt.x;
  o.y = (x.y - mean) * inv * g.y + bt.y;
  o.z = (x.z - mean) * inv * g.z + bt.z;
  o.w = (x.w - mean) * inv * g.w + bt.w;
  *(float4*)&out[row * Dc + tid * 4] = o;
}

extern "C" void kernel_launch(void* const* d_in, const int* in_sizes, int n_in,
                              void* d_out, int out_size, void* d_ws, size_t ws_size,
                              hipStream_t stream) {
  const float* q     = (const float*)d_in[0];
  const float* kv    = (const float*)d_in[1];
  const float* Wq    = (const float*)d_in[2];
  const float* bq    = (const float*)d_in[3];
  const float* Wk    = (const float*)d_in[4];
  const float* bk    = (const float*)d_in[5];
  const float* Wv    = (const float*)d_in[6];
  const float* bv    = (const float*)d_in[7];
  const float* Wo    = (const float*)d_in[8];
  const float* bo    = (const float*)d_in[9];
  const float* gamma = (const float*)d_in[10];
  const float* beta  = (const float*)d_in[11];

  float* out_feat = (float*)d_out;
  float* out_attn = out_feat + (size_t)Bc * Sc * Dc;

  const size_t MSZ = (size_t)Bc * Sc * Ec;          // 8M elements
  __bf16* wsb = (__bf16*)d_ws;
  __bf16* qp   = wsb;                               // [B*S, E] bf16
  __bf16* kp   = wsb + 1 * MSZ;
  __bf16* vp   = wsb + 2 * MSZ;
  __bf16* fctx = wsb + 3 * MSZ;
  float*  fo   = (float*)(wsb + 4 * MSZ);           // [B*S, D] fp32

  const int M = Bc * Sc;                            // 8192
  dim3 gP(M / 128, Ec / 128);                       // 64 x 8

  gemm_bias_kernel<float, true><<<gP, 256, 0, stream>>>(q,  Wq, bq, qp, M, Ec, Dc);
  gemm_bias_kernel<float, true><<<gP, 256, 0, stream>>>(kv, Wk, bk, kp, M, Ec, Dc);
  gemm_bias_kernel<float, true><<<gP, 256, 0, stream>>>(kv, Wv, bv, vp, M, Ec, Dc);

  attn_scores_kernel<<<dim3(Sc / 128, Sc / 128, Bc * Hc), 256, 0, stream>>>(qp, kp, out_attn);
  softmax_kernel<<<dim3(Hc * Bc * Sc), 256, 0, stream>>>(out_attn);
  attn_v_kernel<<<dim3(1, Sc / 128, Bc * Hc), 256, 0, stream>>>(out_attn, vp, fctx);

  gemm_bias_kernel<__bf16, false><<<gP, 256, 0, stream>>>(fctx, Wo, bo, fo, M, Dc, Ec);
  resid_ln_kernel<<<dim3(M), 256, 0, stream>>>(fo, q, gamma, beta, out_feat);
}
